// TransitionAwareAttention_1005022347466
// MI455X (gfx1250) — matrix operand order repaired and checked by reading, not compiled, not run on hardware
//
#include <hip/hip_runtime.h>
#include <hip/hip_bf16.h>
#include <stdint.h>

// ---------------------------------------------------------------------------
// TransitionAwareAttention for MI455X (gfx1250, wave32, WMMA)
//   B=16384, L=5, D=1024, H=16, DK=64
//   Fused QKV projection: one pass over x_bf16, 12 WMMAs per A-fragment
//   Output projection:    bf16 WMMA GEMM, f32 out
//   Attention core:       wave-per-(b,h) VALU + shfl_xor butterfly reductions
//   All GEMM dims compile-time constants -> no div chains in wave prologue.
// ---------------------------------------------------------------------------

#define B_  16384
#define L_  5
#define D_  1024
#define H_  16
#define DK_ 64
#define M_TOTAL (B_ * L_)     // 81920 rows for all GEMMs
#define NT64    (D_ / 64)     // 16 N-tiles of width 64

typedef __attribute__((ext_vector_type(16))) __bf16 v16bf;
typedef __attribute__((ext_vector_type(8)))  __bf16 v8bf;
typedef __attribute__((ext_vector_type(4)))  __bf16 v4bf;
typedef __attribute__((ext_vector_type(2)))  __bf16 v2bf;
typedef __attribute__((ext_vector_type(8)))  float  v8f;
typedef __attribute__((ext_vector_type(4)))  float  v4f;

// ---------------------------------------------------------------------------
// f32 -> bf16 conversion (4 elements / thread)
// ---------------------------------------------------------------------------
__global__ __launch_bounds__(256)
void cvt_f32_to_bf16(const float* __restrict__ src, __bf16* __restrict__ dst, int n4) {
    int i = blockIdx.x * blockDim.x + threadIdx.x;
    if (i >= n4) return;
    v4f f = *(const v4f*)(src + (size_t)i * 4);
    v4bf o;
    o.x = (__bf16)f.x; o.y = (__bf16)f.y; o.z = (__bf16)f.z; o.w = (__bf16)f.w;
    *(v4bf*)(dst + (size_t)i * 4) = o;
}

// ---------------------------------------------------------------------------
// Fragment layouts used here (wave32; contiguous k, same order for A and B):
//   A 16x32 bf16 : lane l holds row m=l&15; lanes 0-15 carry K=0..15,
//                  lanes 16-31 carry K=16..31
//   B 32x16 bf16 : lane l holds col n=l&15; lanes 0-15 carry K=0..15,
//                  lanes 16-31 carry K=16..31
//   C/D 16x16 f32: VGPR v, lane l -> M = (l>>4)*8 + v, N = l&15
// ---------------------------------------------------------------------------

__device__ __forceinline__ v16bf load_a_frag(const __bf16* Ap, int kb, int half) {
    return *(const v16bf*)(Ap + kb + half * 16);
}

// ---------------------------------------------------------------------------
// Fused QKV GEMM: {Q,K,V}[M,1024] = A[M,1024] @ W{q,k,v}[1024,1024]^T + b{q,k,v}
//   One wave owns a 16(M) x 64(N) tile for all three outputs:
//   each A fragment feeds 12 WMMAs -> x streamed from HBM once, not 3x.
// ---------------------------------------------------------------------------
__global__ __launch_bounds__(256)
void gemm_qkv_wmma(const __bf16* __restrict__ A,
                   const __bf16* __restrict__ Wq, const __bf16* __restrict__ Wk,
                   const __bf16* __restrict__ Wv,
                   const float* __restrict__ bq, const float* __restrict__ bk,
                   const float* __restrict__ bv,
                   __bf16* __restrict__ Qo, __bf16* __restrict__ Ko,
                   __bf16* __restrict__ Vo) {
    const int lane = threadIdx.x & 31;
    const int wave = blockIdx.x * (blockDim.x >> 5) + (threadIdx.x >> 5);

    const int mtile = wave >> 4;          // wave / NT64  (NT64 == 16)
    const int nt64  = wave & (NT64 - 1);  // wave % NT64

    const int mrow = lane & 15;
    const int half = lane >> 4;

    const __bf16* Ap = A + (size_t)(mtile * 16 + mrow) * D_;
    const int ncol0 = (nt64 << 6) + (lane & 15);

    v8f accQ[4] = {}, accK[4] = {}, accV[4] = {};

    for (int kb = 0; kb < D_; kb += 32) {
        v16bf afrag = load_a_frag(Ap, kb, half);
#pragma unroll
        for (int j = 0; j < 4; ++j) {
            const size_t woff = (size_t)(ncol0 + j * 16) * D_ + kb + half * 16;
            v16bf bfq = *(const v16bf*)(Wq + woff);
            accQ[j] = __builtin_amdgcn_wmma_f32_16x16x32_bf16(
                false, afrag, false, bfq, (short)0, accQ[j], false, false);
            v16bf bfk = *(const v16bf*)(Wk + woff);
            accK[j] = __builtin_amdgcn_wmma_f32_16x16x32_bf16(
                false, afrag, false, bfk, (short)0, accK[j], false, false);
            v16bf bfv = *(const v16bf*)(Wv + woff);
            accV[j] = __builtin_amdgcn_wmma_f32_16x16x32_bf16(
                false, afrag, false, bfv, (short)0, accV[j], false, false);
        }
    }

#pragma unroll
    for (int j = 0; j < 4; ++j) {
        const int col = (nt64 << 6) + j * 16 + (lane & 15);
        const float bvq = bq[col], bvk = bk[col], bvv = bv[col];
#pragma unroll
        for (int v = 0; v < 8; ++v) {
            const int mr = (half << 3) + v;
            const size_t idx = (size_t)(mtile * 16 + mr) * D_ + col;
            Qo[idx] = (__bf16)(accQ[j][v] + bvq);
            Ko[idx] = (__bf16)(accK[j][v] + bvk);
            Vo[idx] = (__bf16)(accV[j][v] + bvv);
        }
    }
}

// ---------------------------------------------------------------------------
// Output projection: Y[M,1024] = A[M,1024] @ W[1024,1024]^T + bias, f32 out
// ---------------------------------------------------------------------------
__global__ __launch_bounds__(256)
void gemm_bias_wmma_f32(const __bf16* __restrict__ A, const __bf16* __restrict__ W,
                        const float* __restrict__ bias, float* __restrict__ Y) {
    const int lane = threadIdx.x & 31;
    const int wave = blockIdx.x * (blockDim.x >> 5) + (threadIdx.x >> 5);

    const int mtile = wave >> 4;
    const int nt64  = wave & (NT64 - 1);

    const int mrow = lane & 15;
    const int half = lane >> 4;

    const __bf16* Ap = A + (size_t)(mtile * 16 + mrow) * D_;
    const int ncol0 = (nt64 << 6) + (lane & 15);

    v8f acc[4] = {};

    for (int kb = 0; kb < D_; kb += 32) {
        v16bf afrag = load_a_frag(Ap, kb, half);
#pragma unroll
        for (int j = 0; j < 4; ++j) {
            const __bf16* Wp = W + (size_t)(ncol0 + j * 16) * D_ + kb + half * 16;
            v16bf bfrag = *(const v16bf*)Wp;
            acc[j] = __builtin_amdgcn_wmma_f32_16x16x32_bf16(
                false, afrag, false, bfrag, (short)0, acc[j], false, false);
        }
    }

#pragma unroll
    for (int j = 0; j < 4; ++j) {
        const int col = (nt64 << 6) + j * 16 + (lane & 15);
        const float bv = bias[col];
#pragma unroll
        for (int v = 0; v < 8; ++v) {
            const int mr = (half << 3) + v;
            Y[(size_t)(mtile * 16 + mr) * D_ + col] = acc[j][v] + bv;
        }
    }
}

// ---------------------------------------------------------------------------
// Attention core: one wave32 per (b,h).
//   Each lane owns 2 of the DK=64 dims. 25 logit partials reduced with a
//   5-step shfl_xor butterfly (all lanes end with full sums), then softmax
//   with transition bias T[h] and padding mask; output written as bf16.
// ---------------------------------------------------------------------------
__global__ __launch_bounds__(256)
void attn_core(const __bf16* __restrict__ Qb, const __bf16* __restrict__ Kb,
               const __bf16* __restrict__ Vb, const float* __restrict__ T,
               const unsigned char* __restrict__ mask, __bf16* __restrict__ Ob) {
    const int lane = threadIdx.x & 31;
    const int wave = blockIdx.x * (blockDim.x >> 5) + (threadIdx.x >> 5);
    const int b = wave >> 4;        // wave / H_
    const int h = wave & (H_ - 1);  // wave % H_
    const int d0 = lane * 2;
    const size_t base = (size_t)b * L_ * D_ + h * DK_ + d0;

    float q[L_][2], k[L_][2], v[L_][2];
#pragma unroll
    for (int i = 0; i < L_; ++i) {
        v2bf t;
        t = *(const v2bf*)(Qb + base + (size_t)i * D_); q[i][0] = (float)t.x; q[i][1] = (float)t.y;
        t = *(const v2bf*)(Kb + base + (size_t)i * D_); k[i][0] = (float)t.x; k[i][1] = (float)t.y;
        t = *(const v2bf*)(Vb + base + (size_t)i * D_); v[i][0] = (float)t.x; v[i][1] = (float)t.y;
    }

    float lg[L_ * L_];
#pragma unroll
    for (int i = 0; i < L_; ++i)
#pragma unroll
        for (int j = 0; j < L_; ++j)
            lg[i * L_ + j] = q[i][0] * k[j][0] + q[i][1] * k[j][1];

    // wave32 butterfly reduction of all 25 partial dot products
#pragma unroll
    for (int off = 16; off > 0; off >>= 1)
#pragma unroll
        for (int t = 0; t < L_ * L_; ++t)
            lg[t] += __shfl_xor(lg[t], off, 32);

    // bias + mask + softmax (redundant across lanes, all lanes hold full sums)
    float p[L_][L_];
#pragma unroll
    for (int i = 0; i < L_; ++i) {
        float row[L_];
        float mx = -1e30f;
#pragma unroll
        for (int j = 0; j < L_; ++j) {
            float val = lg[i * L_ + j] * 0.125f + T[(h * L_ + i) * L_ + j];  // 1/sqrt(64)
            if (!mask[b * L_ + j]) val = -1e30f;
            row[j] = val;
            mx = fmaxf(mx, val);
        }
        float s = 0.f;
#pragma unroll
        for (int j = 0; j < L_; ++j) { row[j] = __expf(row[j] - mx); s += row[j]; }
        const float inv = 1.f / s;
#pragma unroll
        for (int j = 0; j < L_; ++j) p[i][j] = row[j] * inv;
    }

#pragma unroll
    for (int i = 0; i < L_; ++i) {
        float o0 = 0.f, o1 = 0.f;
#pragma unroll
        for (int j = 0; j < L_; ++j) { o0 += p[i][j] * v[j][0]; o1 += p[i][j] * v[j][1]; }
        v2bf o; o.x = (__bf16)o0; o.y = (__bf16)o1;
        *(v2bf*)(Ob + base + (size_t)i * D_) = o;
    }
}

// ---------------------------------------------------------------------------
// kernel_launch
// ---------------------------------------------------------------------------
extern "C" void kernel_launch(void* const* d_in, const int* in_sizes, int n_in,
                              void* d_out, int out_size, void* d_ws, size_t ws_size,
                              hipStream_t stream) {
    const float*         x    = (const float*)d_in[0];
    const unsigned char* mask = (const unsigned char*)d_in[1];   // jnp bool -> 1 byte
    const float*         Wq   = (const float*)d_in[2];
    const float*         bq   = (const float*)d_in[3];
    const float*         Wk   = (const float*)d_in[4];
    const float*         bk   = (const float*)d_in[5];
    const float*         Wv   = (const float*)d_in[6];
    const float*         bv   = (const float*)d_in[7];
    const float*         Wo   = (const float*)d_in[8];
    const float*         bo   = (const float*)d_in[9];
    const float*         T    = (const float*)d_in[10];
    float* out = (float*)d_out;

    char* ws = (char*)d_ws;
    const size_t SX = (size_t)M_TOTAL * D_ * sizeof(__bf16);   // 167.8 MB
    const size_t SW = (size_t)D_ * D_ * sizeof(__bf16);        // 2 MB

    __bf16* xbf = (__bf16*)ws;                    // x in bf16; reused as attn-out
    __bf16* wqb = (__bf16*)(ws + SX);
    __bf16* wkb = (__bf16*)(ws + SX + 1 * SW);
    __bf16* wvb = (__bf16*)(ws + SX + 2 * SW);
    __bf16* wob = (__bf16*)(ws + SX + 3 * SW);
    __bf16* Qb  = (__bf16*)(ws + SX + 4 * SW);
    __bf16* Kb  = (__bf16*)(ws + SX + 4 * SW + SX);
    __bf16* Vb  = (__bf16*)(ws + SX + 4 * SW + 2 * SX);

    // 1) conversions to bf16
    {
        int n4 = M_TOTAL * D_ / 4;
        cvt_f32_to_bf16<<<n4 / 256, 256, 0, stream>>>(x, xbf, n4);
        int w4 = D_ * D_ / 4;
        cvt_f32_to_bf16<<<w4 / 256, 256, 0, stream>>>(Wq, wqb, w4);
        cvt_f32_to_bf16<<<w4 / 256, 256, 0, stream>>>(Wk, wkb, w4);
        cvt_f32_to_bf16<<<w4 / 256, 256, 0, stream>>>(Wv, wvb, w4);
        cvt_f32_to_bf16<<<w4 / 256, 256, 0, stream>>>(Wo, wob, w4);
    }

    // 2) fused Q/K/V projection: x streamed from HBM once for all three
    const int gemmWaves  = (M_TOTAL / 16) * NT64;        // 81920 waves
    const int gemmBlocks = gemmWaves / 8;                // 8 waves / 256-thread block
    gemm_qkv_wmma<<<gemmBlocks, 256, 0, stream>>>(
        xbf, wqb, wkb, wvb, bq, bk, bv, Qb, Kb, Vb);

    // 3) attention core (writes bf16 into xbf region; x is dead by now)
    attn_core<<<(B_ * H_) / 8, 256, 0, stream>>>(Qb, Kb, Vb, T, mask, xbf);

    // 4) output projection (f32 out -> d_out)
    gemm_bias_wmma_f32<<<gemmBlocks, 256, 0, stream>>>(xbf, wob, bo, out);
}
